// TGMM_24129126269446
// MI455X (gfx1250) — matrix operand order repaired and checked by reading, not compiled, not run on hardware
//
#include <hip/hip_runtime.h>
#include <hip/hip_bf16.h>

typedef __attribute__((ext_vector_type(16))) __bf16 v16bf;
typedef __attribute__((ext_vector_type(8)))  __bf16 v8bf;
typedef __attribute__((ext_vector_type(8)))  float  v8f;

#define HW    3136
#define B_    16
#define CIN   512
#define NL    20
#define KCc   256
#define VCc   512
#define NHEAD 8
#define DK    32
#define DV    64
#define LIN   768

__device__ __forceinline__ __bf16 f2bf(float f) { return (__bf16)f; }

// Fragment-major LDS layout (wave32 WMMA 16x16x32 bf16):
//  A (16x32 MxK):  element (m,k) -> lane = m + 16*((k>>3)&1), slot = (k>>4)*8 + (k&7)
//  B (32x16 KxN):  element (k,n) -> lane = (n&15) + 16*((k>>3)&1), slot = (k>>4)*8 + (k&7)
// Each lane's 16 bf16 fragment is contiguous (32B, aligned) -> 2x ds_load_b128.

// ---------------------------------------------------------------------------
// Main GEMM over pixels: Y[b,o,hw] = act(sum_c W[o,c] * Xin[b,c,hw] + bias[o])
// FUSED: Xin := (w1out - mean[b,c]) * rstd[b,c] * vis[b,c,hw]  (final pm GEMM)
// Block tile: 128(M) x 64(N), K-step 32, double-buffered LDS (1 barrier/step);
// 8 waves, each one 16x64 strip = 4 WMMAs/step with A-fragment reuse.
// ---------------------------------------------------------------------------
template<bool RELU, bool FUSED>
__global__ __launch_bounds__(256)
void gemm512_kernel(const float* __restrict__ W, const float* __restrict__ bias,
                    const float* __restrict__ X, const float* __restrict__ visp,
                    const float* __restrict__ chmean, const float* __restrict__ chrstd,
                    float* __restrict__ Y, int O)
{
  __shared__ __bf16 As[2][8 * 32 * 16];   // ping-pong: 8 m-tiles x 32 lane-slots x 16
  __shared__ __bf16 Bs[2][4 * 32 * 16];   // ping-pong: 4 n-tiles x 32 lane-slots x 16
  const int tid  = threadIdx.x;
  const int wave = tid >> 5;
  const int lane = tid & 31;
  const int m0   = blockIdx.y * 128;
  const int n0   = blockIdx.x * 64;
  const int b    = blockIdx.z;
  const size_t xbase = (size_t)b * CIN * HW;

  // A staging: thread -> (row, half): 16 consecutive k of one row -> 2 lanes x 8 slots.
  const int a_row  = tid >> 1;
  const int a_half = tid & 1;
  const int a_off_lo = (((a_row >> 4) * 32 + (a_row & 15)     ) * 16) + a_half * 8;
  const int a_off_hi = (((a_row >> 4) * 32 + (a_row & 15) + 16) * 16) + a_half * 8;
  // B staging: thread -> (col nn 0..63, kq 0..3): 8 consecutive k of one column.
  const int b_nn = tid & 63;
  const int b_kq = tid >> 6;                                   // k = b_kq*8 + j
  const int b_l  = (b_nn & 15) + ((b_kq & 1) << 4);            // 16*((k>>3)&1)
  const int b_off = (((b_nn >> 4) * 32 + b_l) * 16) + (b_kq >> 1) * 8;

  auto stageA = [&](int k0, int p) {
    const float4* s4 = (const float4*)&W[(size_t)(m0 + a_row) * CIN + k0 + a_half * 16];
    const float4 f0 = s4[0], f1 = s4[1], f2 = s4[2], f3 = s4[3];
    v8bf lo, hi;
    lo[0] = f2bf(f0.x); lo[1] = f2bf(f0.y); lo[2] = f2bf(f0.z); lo[3] = f2bf(f0.w);
    lo[4] = f2bf(f1.x); lo[5] = f2bf(f1.y); lo[6] = f2bf(f1.z); lo[7] = f2bf(f1.w);
    hi[0] = f2bf(f2.x); hi[1] = f2bf(f2.y); hi[2] = f2bf(f2.z); hi[3] = f2bf(f2.w);
    hi[4] = f2bf(f3.x); hi[5] = f2bf(f3.y); hi[6] = f2bf(f3.z); hi[7] = f2bf(f3.w);
    *(v8bf*)&As[p][a_off_lo] = lo;   // k = half*16 + 0..7
    *(v8bf*)&As[p][a_off_hi] = hi;   // k = half*16 + 8..15
  };
  auto stageB = [&](int k0, int p) {
    v8bf q;
#pragma unroll
    for (int j = 0; j < 8; ++j) {
      const int c = k0 + b_kq * 8 + j;
      const size_t gi = xbase + (size_t)c * HW + n0 + b_nn;
      float xv = X[gi];
      if (FUSED) {
        xv = (xv - chmean[b * CIN + c]) * chrstd[b * CIN + c] * visp[gi];
      }
      q[j] = f2bf(xv);
    }
    *(v8bf*)&Bs[p][b_off] = q;
  };

  v8f acc[4];
  acc[0] = v8f{}; acc[1] = v8f{}; acc[2] = v8f{}; acc[3] = v8f{};

  stageA(0, 0);
  stageB(0, 0);
  int p = 0;
  for (int k0 = 0; k0 < CIN; k0 += 32) {
    __syncthreads();                         // staged tile p is visible
    if (k0 + 32 < CIN) {                     // overlap: stage next tile into p^1
      stageA(k0 + 32, p ^ 1);
      stageB(k0 + 32, p ^ 1);
      if (k0 + 64 < CIN) {                   // hint tile after next (global_prefetch_b8)
        __builtin_prefetch(&W[(size_t)(m0 + a_row) * CIN + k0 + 64], 0, 1);
        __builtin_prefetch(&X[xbase + (size_t)(k0 + 64 + b_kq * 8) * HW + n0 + b_nn], 0, 1);
      }
    }
    const v16bf a = *(const v16bf*)&As[p][(wave * 32 + lane) * 16];
#pragma unroll
    for (int t = 0; t < 4; ++t) {
      const v16bf bfr = *(const v16bf*)&Bs[p][(t * 32 + lane) * 16];
      acc[t] = __builtin_amdgcn_wmma_f32_16x16x32_bf16(false, a, false, bfr, (short)0, acc[t], false, false);
    }
    p ^= 1;
  }

#pragma unroll
  for (int t = 0; t < 4; ++t) {
#pragma unroll
    for (int r = 0; r < 8; ++r) {
      const int ml = 16 * wave + r + ((lane >> 4) << 3);
      const int nl = (lane & 15) + 16 * t;
      const int o  = m0 + ml;
      float v = acc[t][r] + bias[o];
      if (RELU) v = fmaxf(v, 0.f);
      Y[((size_t)b * O + o) * HW + n0 + nl] = v;
    }
  }
}

// Per-(b,channel) mean / rstd over HW (instance norm stats, population var)
__global__ __launch_bounds__(256)
void stats_kernel(const float* __restrict__ X, float* __restrict__ mean, float* __restrict__ rstd)
{
  __shared__ float ss[256], ss2[256];
  const int bc = blockIdx.x;
  const float* p = X + (size_t)bc * HW;
  float s = 0.f, s2 = 0.f;
  for (int i = threadIdx.x; i < HW; i += 256) { const float v = p[i]; s += v; s2 += v * v; }
  ss[threadIdx.x] = s; ss2[threadIdx.x] = s2;
  __syncthreads();
  for (int o = 128; o > 0; o >>= 1) {
    if (threadIdx.x < o) { ss[threadIdx.x] += ss[threadIdx.x + o]; ss2[threadIdx.x] += ss2[threadIdx.x + o]; }
    __syncthreads();
  }
  if (threadIdx.x == 0) {
    const float mu  = ss[0] * (1.f / HW);
    const float var = ss2[0] * (1.f / HW) - mu * mu;
    mean[bc] = mu;
    rstd[bc] = rsqrtf(var + 1e-5f);
  }
}

// ---------------------------------------------------------------------------
// Attention: per (b, head), 4 waves/block, each wave one 16-row HW tile.
// sim = (inorm(q) @ k) * KC^-0.5 + mask_bias ; softmax(n<20) ; pool += sim
// out = sim @ v^T  -> attn_out[b, h*64+d, hw]
// ---------------------------------------------------------------------------
__global__ __launch_bounds__(128)
void attn_kernel(const float* __restrict__ qraw, const float* __restrict__ qmean,
                 const float* __restrict__ qrstd,
                 const float* __restrict__ keyp, const float* __restrict__ valp,
                 const float* __restrict__ lmask,
                 float* __restrict__ attn_out, float* __restrict__ poolsum)
{
  __shared__ __bf16 kBf[2 * 32 * 16];      // B-frag layout: K=dk(32) x N=n(32), n>=20 zero
  __shared__ __bf16 vBf[4 * 32 * 16];      // B-frag layout: K=n(32)  x N=d(64), n>=20 zero
  __shared__ float  sbias[32];
  __shared__ float  simS[4][16][32];
  __shared__ __bf16 simBf[4 * 32 * 16];    // A-frag layout per wave: M=hw(16) x K=n(32)
  __shared__ float  poolAcc[NL];

  const int tid  = threadIdx.x;
  const int wave = tid >> 5;
  const int lane = tid & 31;
  const int hd   = blockIdx.y;
  const int b    = blockIdx.z;
  const int hw0  = (blockIdx.x * 4 + wave) * 16;

  // kB staging: thread -> (n, kq8): 8 consecutive k, fixed n -> 1 lane x 8 slots
  {
    const int nn  = tid & 31;
    const int kq8 = tid >> 5;                         // 0..3, k = kq8*8 + i
    const int l   = (nn & 15) + ((kq8 & 1) << 4);     // 16*((k>>3)&1)
    v8bf q;
#pragma unroll
    for (int i = 0; i < 8; ++i) {
      const int k = kq8 * 8 + i;
      float v = 0.f;
      if (nn < NL) v = keyp[((size_t)b * KCc + hd * DK + k) * NL + nn];
      q[i] = f2bf(v);
    }
    *(v8bf*)&kBf[(((nn >> 4) * 32 + l) * 16) + (kq8 >> 1) * 8] = q;
  }
  // vB staging: thread -> (d, half): 16 consecutive n, fixed d -> 1 lane x 16 slots
  {
    const int d    = tid & 63;
    const int half = tid >> 6;                        // n = half*16 + i
    const int l    = (d & 15) + (half << 4);
    v16bf q;
#pragma unroll
    for (int i = 0; i < 16; ++i) {
      const int n = half * 16 + i;
      float v = 0.f;
      if (n < NL) v = valp[((size_t)b * VCc + hd * DV + d) * NL + n];
      q[i] = f2bf(v);
    }
    *(v16bf*)&vBf[((d >> 4) * 32 + l) * 16] = q;
  }
  if (tid < 32) {
    float bias = -30000.f;
    if (tid < NL) bias = 10000.f * lmask[b * NL + tid] - 10000.f;
    sbias[tid] = bias;
  }
  if (tid < NL) poolAcc[tid] = 0.f;
  __syncthreads();

  // Q fragment from global, instance-norm applied on the fly
  v16bf qa;
  {
    const int m  = lane & 15;
    const int kh = (lane >> 4) << 3;
#pragma unroll
    for (int v = 0; v < 8; ++v) {
      const int kb = ((v >> 2) << 4) + kh + ((v & 3) << 1);
#pragma unroll
      for (int j = 0; j < 2; ++j) {
        const int c = hd * DK + kb + j;
        float q = qraw[((size_t)b * KCc + c) * HW + hw0 + m];
        q = (q - qmean[b * KCc + c]) * qrstd[b * KCc + c];
        qa[2 * v + j] = f2bf(q);
      }
    }
  }
  const v16bf kb0 = *(const v16bf*)&kBf[(lane) * 16];
  const v16bf kb1 = *(const v16bf*)&kBf[(32 + lane) * 16];
  v8f s0 = {}, s1 = {};
  s0 = __builtin_amdgcn_wmma_f32_16x16x32_bf16(false, qa, false, kb0, (short)0, s0, false, false);
  s1 = __builtin_amdgcn_wmma_f32_16x16x32_bf16(false, qa, false, kb1, (short)0, s1, false, false);

  const float scale = 0.0625f;  // 256^-0.5
#pragma unroll
  for (int r = 0; r < 8; ++r) {
    const int m = r + ((lane >> 4) << 3);
    simS[wave][m][lane & 15]        = s0[r] * scale;
    simS[wave][m][(lane & 15) + 16] = s1[r] * scale;
  }
  __syncthreads();

  if (lane < 16) {            // one row per lane: 20-wide softmax
    const int m = lane;
    float p[NL];
    float mx = -1e30f;
#pragma unroll
    for (int n = 0; n < NL; ++n) { p[n] = simS[wave][m][n] + sbias[n]; mx = fmaxf(mx, p[n]); }
    float sum = 0.f;
#pragma unroll
    for (int n = 0; n < NL; ++n) { p[n] = __expf(p[n] - mx); sum += p[n]; }
    const float inv = 1.f / sum;
    float sm[32];
#pragma unroll
    for (int n = 0; n < NL; ++n) { sm[n] = p[n] * inv; atomicAdd(&poolAcc[n], sm[n]); }
#pragma unroll
    for (int n = NL; n < 32; ++n) sm[n] = 0.f;
    // A-frag layout: lane-slot m holds k = 0..15 ; slot m+16 holds 16..31
    v16bf vlo, vhi;
#pragma unroll
    for (int i = 0; i < 8; ++i) {
      vlo[i]     = f2bf(sm[i]);        // k = 0..7
      vlo[8 + i] = f2bf(sm[8 + i]);    // k = 8..15
      vhi[i]     = f2bf(sm[16 + i]);   // k = 16..23
      vhi[8 + i] = f2bf(sm[24 + i]);   // k = 24..31
    }
    *(v16bf*)&simBf[(wave * 32 + m) * 16]      = vlo;
    *(v16bf*)&simBf[(wave * 32 + m + 16) * 16] = vhi;
  }
  __syncthreads();

  if (tid < NL) atomicAdd(&poolsum[((size_t)b * NHEAD + hd) * NL + tid], poolAcc[tid]);

  const v16bf sa = *(const v16bf*)&simBf[(wave * 32 + lane) * 16];
  v8f o0 = {}, o1 = {}, o2 = {}, o3 = {};
  {
    v16bf vb;
    vb = *(const v16bf*)&vBf[(0 * 32 + lane) * 16];
    o0 = __builtin_amdgcn_wmma_f32_16x16x32_bf16(false, sa, false, vb, (short)0, o0, false, false);
    vb = *(const v16bf*)&vBf[(1 * 32 + lane) * 16];
    o1 = __builtin_amdgcn_wmma_f32_16x16x32_bf16(false, sa, false, vb, (short)0, o1, false, false);
    vb = *(const v16bf*)&vBf[(2 * 32 + lane) * 16];
    o2 = __builtin_amdgcn_wmma_f32_16x16x32_bf16(false, sa, false, vb, (short)0, o2, false, false);
    vb = *(const v16bf*)&vBf[(3 * 32 + lane) * 16];
    o3 = __builtin_amdgcn_wmma_f32_16x16x32_bf16(false, sa, false, vb, (short)0, o3, false, false);
  }
  const v8f outs[4] = {o0, o1, o2, o3};
#pragma unroll
  for (int t = 0; t < 4; ++t) {
#pragma unroll
    for (int r = 0; r < 8; ++r) {
      const int m = r + ((lane >> 4) << 3);
      const int d = 16 * t + (lane & 15);
      attn_out[((size_t)b * VCc + hd * DV + d) * HW + hw0 + m] = outs[t][r];
    }
  }
}

// ----------------------------- small kernels -------------------------------
__global__ void zero_kernel(float* p, int n) {
  const int i = blockIdx.x * blockDim.x + threadIdx.x;
  if (i < n) p[i] = 0.f;
}

// out[b,o,n] = (bias[o] + sum_c w[o,c] * l[b,c,n]) * lmask[b,n]   (C = 768)
__global__ void lang_proj_kernel(const float* __restrict__ w, const float* __restrict__ bias,
                                 const float* __restrict__ l, const float* __restrict__ lmask,
                                 float* __restrict__ out, int O)
{
  const int idx = blockIdx.x * blockDim.x + threadIdx.x;
  if (idx >= B_ * O * NL) return;
  const int n = idx % NL, o = (idx / NL) % O, b = idx / (NL * O);
  float s = bias[o];
  const float* wr = w + (size_t)o * LIN;
  const float* lb = l + (size_t)b * LIN * NL + n;
  for (int c = 0; c < LIN; ++c) s += wr[c] * lb[c * NL];
  out[idx] = s * lmask[b * NL + n];
}

// l_new[b,vc,n] = (poolsum[b,h,n]/HW) * value2[b,vc,n], h = vc/64
__global__ void lnew_kernel(const float* __restrict__ poolsum, const float* __restrict__ value2,
                            float* __restrict__ l_new)
{
  const int idx = blockIdx.x * blockDim.x + threadIdx.x;
  if (idx >= B_ * VCc * NL) return;
  const int n = idx % NL, vc = (idx / NL) % VCc, b = idx / (NL * VCc);
  const int h = vc >> 6;
  l_new[idx] = poolsum[((size_t)b * NHEAD + h) * NL + n] * (1.f / HW) * value2[idx];
}

// out[b,o,n] = bias[o] + sum_c w[o,c] * x[b,c,n]   (C = 512)
__global__ void proj512_kernel(const float* __restrict__ w, const float* __restrict__ bias,
                               const float* __restrict__ x, float* __restrict__ out)
{
  const int idx = blockIdx.x * blockDim.x + threadIdx.x;
  if (idx >= B_ * VCc * NL) return;
  const int n = idx % NL, o = (idx / NL) % VCc, b = idx / (NL * VCc);
  float s = bias[o];
  const float* wr = w + (size_t)o * VCc;
  const float* xb = x + (size_t)b * VCc * NL + n;
  for (int c = 0; c < VCc; ++c) s += wr[c] * xb[c * NL];
  out[idx] = s;
}

__global__ void embp_kernel(const float* __restrict__ pe_w, const float* __restrict__ pe_b,
                            const float* __restrict__ emb, float* __restrict__ embp)
{
  const int idx = blockIdx.x * blockDim.x + threadIdx.x;
  if (idx >= B_ * VCc) return;
  const int o = idx % VCc, b = idx / VCc;
  float s = pe_b[o];
  const float* wr = pe_w + (size_t)o * LIN;
  const float* eb = emb + (size_t)b * LIN;
  for (int c = 0; c < LIN; ++c) s += wr[c] * eb[c];
  embp[idx] = s;
}

__global__ void lnm_kernel(const float* __restrict__ l_new, float* __restrict__ lnm)
{
  const int idx = blockIdx.x * blockDim.x + threadIdx.x;
  if (idx >= B_ * VCc) return;
  float s = 0.f;
  for (int n = 0; n < NL; ++n) s += l_new[(size_t)idx * NL + n];
  lnm[idx] = s * (1.f / NL);
}

// l_out[b,o] = w3_b[o] + sum_c w3[o,c]*(lnm[b,c]*gamma + embp[b,c]) + embp[b,o]
__global__ void lout_kernel(const float* __restrict__ w3_w, const float* __restrict__ w3_b,
                            const float* __restrict__ lnm, const float* __restrict__ embp,
                            const float* __restrict__ gamma, float* __restrict__ lout)
{
  const int idx = blockIdx.x * blockDim.x + threadIdx.x;
  if (idx >= B_ * VCc) return;
  const int o = idx % VCc, b = idx / VCc;
  const float g = gamma[0];
  float s = w3_b[o];
  const float* wr = w3_w + (size_t)o * VCc;
  for (int c = 0; c < VCc; ++c) s += wr[c] * (lnm[b * VCc + c] * g + embp[b * VCc + c]);
  lout[idx] = s + embp[idx];
}

// ---------------------------------------------------------------------------
extern "C" void kernel_launch(void* const* d_in, const int* in_sizes, int n_in,
                              void* d_out, int out_size, void* d_ws, size_t ws_size,
                              hipStream_t stream)
{
  const float* x      = (const float*)d_in[0];
  const float* l      = (const float*)d_in[1];
  const float* l_mask = (const float*)d_in[2];
  const float* emb    = (const float*)d_in[3];
  const float* vis_w  = (const float*)d_in[4];
  const float* vis_b  = (const float*)d_in[5];
  const float* fq_w   = (const float*)d_in[6];
  const float* fq_b   = (const float*)d_in[7];
  const float* fk_w   = (const float*)d_in[8];
  const float* fk_b   = (const float*)d_in[9];
  const float* fv_w   = (const float*)d_in[10];
  const float* fv_b   = (const float*)d_in[11];
  const float* fv2_w  = (const float*)d_in[12];
  const float* fv2_b  = (const float*)d_in[13];
  const float* w1_w   = (const float*)d_in[14];
  const float* w1_b   = (const float*)d_in[15];
  const float* w2_w   = (const float*)d_in[16];
  const float* w2_b   = (const float*)d_in[17];
  const float* w3_w   = (const float*)d_in[18];
  const float* w3_b   = (const float*)d_in[19];
  const float* pe_w   = (const float*)d_in[20];
  const float* pe_b   = (const float*)d_in[21];
  const float* pm_w   = (const float*)d_in[22];
  const float* pm_b   = (const float*)d_in[23];
  const float* gamma  = (const float*)d_in[24];
  (void)in_sizes; (void)n_in; (void)out_size; (void)ws_size;

  float* ws = (float*)d_ws;
  size_t off = 0;
  auto alloc = [&](size_t n) { float* p = ws + off; off += n; return p; };
  float* vis      = alloc((size_t)B_ * VCc * HW);
  float* qraw     = alloc((size_t)B_ * KCc * HW);
  float* attn_out = alloc((size_t)B_ * VCc * HW);
  float* w1out    = alloc((size_t)B_ * VCc * HW);
  float* qmean    = alloc(B_ * KCc);
  float* qrstd    = alloc(B_ * KCc);
  float* w1mean   = alloc(B_ * VCc);
  float* w1rstd   = alloc(B_ * VCc);
  float* keyb     = alloc((size_t)B_ * KCc * NL);
  float* valb     = alloc((size_t)B_ * VCc * NL);
  float* val2b    = alloc((size_t)B_ * VCc * NL);
  float* poolsum  = alloc(B_ * NHEAD * NL);
  float* l_new    = alloc((size_t)B_ * VCc * NL);
  float* embp     = alloc(B_ * VCc);
  float* lnm      = alloc(B_ * VCc);

  float* mm_out    = (float*)d_out;
  float* lout_out  = mm_out + (size_t)B_ * VCc * HW;
  float* lnew2_out = lout_out + B_ * VCc;

  // Language projections (tiny)
  {
    const int nk = B_ * KCc * NL;
    lang_proj_kernel<<<(nk + 255) / 256, 256, 0, stream>>>(fk_w, fk_b, l, l_mask, keyb, KCc);
    const int nv = B_ * VCc * NL;
    lang_proj_kernel<<<(nv + 255) / 256, 256, 0, stream>>>(fv_w, fv_b, l, l_mask, valb, VCc);
    lang_proj_kernel<<<(nv + 255) / 256, 256, 0, stream>>>(fv2_w, fv2_b, l, l_mask, val2b, VCc);
  }

  // Vision GEMMs (WMMA bf16, 128x64 block tile, double-buffered)
  gemm512_kernel<true,  false><<<dim3(HW / 64, VCc / 128, B_), 256, 0, stream>>>(
      vis_w, vis_b, x, nullptr, nullptr, nullptr, vis, VCc);
  gemm512_kernel<false, false><<<dim3(HW / 64, KCc / 128, B_), 256, 0, stream>>>(
      fq_w, fq_b, x, nullptr, nullptr, nullptr, qraw, KCc);
  stats_kernel<<<B_ * KCc, 256, 0, stream>>>(qraw, qmean, qrstd);

  zero_kernel<<<(B_ * NHEAD * NL + 255) / 256, 256, 0, stream>>>(poolsum, B_ * NHEAD * NL);
  attn_kernel<<<dim3(HW / 64, NHEAD, B_), 128, 0, stream>>>(
      qraw, qmean, qrstd, keyb, valb, l_mask, attn_out, poolsum);

  gemm512_kernel<false, false><<<dim3(HW / 64, VCc / 128, B_), 256, 0, stream>>>(
      w1_w, w1_b, attn_out, nullptr, nullptr, nullptr, w1out, VCc);
  stats_kernel<<<B_ * VCc, 256, 0, stream>>>(w1out, w1mean, w1rstd);

  // mm = relu(pm_w @ (vis * inorm(w1out)) + pm_b)  -- norm+mul fused in staging
  gemm512_kernel<true, true><<<dim3(HW / 64, VCc / 128, B_), 256, 0, stream>>>(
      pm_w, pm_b, w1out, vis, w1mean, w1rstd, mm_out, VCc);

  // Language-side outputs
  {
    const int nl = B_ * VCc * NL;
    lnew_kernel<<<(nl + 255) / 256, 256, 0, stream>>>(poolsum, val2b, l_new);
    proj512_kernel<<<(nl + 255) / 256, 256, 0, stream>>>(w2_w, w2_b, l_new, lnew2_out);
    const int nb = B_ * VCc;
    embp_kernel<<<(nb + 255) / 256, 256, 0, stream>>>(pe_w, pe_b, emb, embp);
    lnm_kernel<<<(nb + 255) / 256, 256, 0, stream>>>(l_new, lnm);
    lout_kernel<<<(nb + 255) / 256, 256, 0, stream>>>(w3_w, w3_b, lnm, embp, gamma, lout_out);
  }
}
